// GEFA_HGAN_19292993093686
// MI455X (gfx1250) — matrix operand order repaired and checked by reading, not compiled, not run on hardware
//
#include <hip/hip_runtime.h>

// ============================================================================
// GEFA-HGAN forward for MI455X (gfx1250, wave32, WMMA).
//
// Heavy GEMMs (meta-path projections X@Wg[m], semantic z@sw1) run as bf16
// v_wmma_f32_16x16x32_bf16 with f32 accumulation; each wave computes a 64x32
// C macro-tile (4 A-frags x 2 B-frags -> 8 accumulators) to raise arithmetic
// intensity and cut weight-panel re-reads 4x vs a 16x32 tile. Graph
// scatter/softmax and GCN aggregation are atomic-based kernels; tiny tail
// MLPs are plain f32.
//
// Input flattening assumption (jax pytree order): top-level dict in insertion
// order, then params dict keys sorted alphabetically with hgan list in order:
//  0 drug_x  1 drug_edge_index  2 drug_batch  3 prot_x  4 prot_edge_index
//  5 prot_edge_attr  6 prot_lens
//  7 c1_b 8 c1_w 9 c2_b 10 c2_w 11 fc1_b 12 fc1_w 13 fc2_b 14 fc2_w
// 15 fc_g1_b 16 fc_g1_w 17 fg1t_b 18 fg1t_w 19 fg2t_b 20 fg2t_w 21 fl_b 22 fl_w
// 23+7l: hgan[l]: Wg, al, ar, bg, sb1, sw1, sw2   (l = 0..3)
// 51 out_b 52 out_w 53 rb_gb 54 rb_gw 55 rb_lw 56 sl_b 57 sl_w
// ============================================================================

typedef unsigned short u16;
typedef __attribute__((ext_vector_type(16))) __bf16 v16bf;
typedef __attribute__((ext_vector_type(8)))  float  v8f;

#define DEV static __device__ __forceinline__

#define BB 64
#define LL 512
#define NATM 32
#define ND 2048            // drug nodes
#define EDRUG 8192         // drug edges
#define MM 64              // meta-paths
#define DHEAD 64           // per-head out size (2*latent)
#define NPROT 32832        // B*(L+1)
#define EPROT 589824       // protein edges
#define PSTRIDE 513        // rows per protein graph

// ---------------------------------------------------------------------------
DEV u16 f2bf(float f) {
  unsigned u = __float_as_uint(f);
  if ((u & 0x7F800000u) == 0x7F800000u) return (u16)(u >> 16); // inf/nan
  unsigned r = u + 0x7FFFu + ((u >> 16) & 1u);                 // RNE
  return (u16)(r >> 16);
}

DEV void atomicMaxF(float* addr, float val) {
  unsigned* ua = (unsigned*)addr;
  unsigned old = *ua;
  for (;;) {
    if (__uint_as_float(old) >= val) return;
    unsigned assumed = old;
    old = atomicCAS(ua, assumed, __float_as_uint(val));
    if (old == assumed) return;
  }
}

// ---------------------------------------------------------------------------
__global__ void k_fill(float* __restrict__ p, float v, size_t n) {
  size_t i = (size_t)blockIdx.x * blockDim.x + threadIdx.x;
  if (i < n) p[i] = v;
}

__global__ void k_f2bf(const float* __restrict__ s, u16* __restrict__ d, size_t n) {
  size_t i = (size_t)blockIdx.x * blockDim.x + threadIdx.x;
  if (i < n) d[i] = f2bf(s[i]);
}

// W [K,N] f32 -> Wt [N,K] bf16, batched over blockIdx.z (stride K*N both sides)
__global__ void k_transpose_bf(const float* __restrict__ W, u16* __restrict__ Wt,
                               int K, int N) {
  size_t base = (size_t)blockIdx.z * K * N;
  size_t i = (size_t)blockIdx.x * blockDim.x + threadIdx.x;
  if (i >= (size_t)K * N) return;
  int n = (int)(i / K), k = (int)(i - (size_t)n * K);
  Wt[base + (size_t)n * K + k] = f2bf(W[base + (size_t)k * N + n]);
}

// ---------------------------------------------------------------------------
// C[R,N] (f32) = A[R,K] (bf16 row-major) * Bt[N,K]^T (bf16, pre-transposed).
// One wave per block; each wave computes a 64x32 C macro-tile: 4 A fragments
// (rows) x 2 B fragments (cols) -> 8 v_wmma_f32_16x16x32_bf16 per k-step.
// R must be a multiple of 64, N of 32, K of 32. Batched via blockIdx.z.
struct alignas(16) Q4 { unsigned x, y, z, w; };
union Frag { Q4 q[2]; v16bf v; };

__global__ void __launch_bounds__(32)
k_wmma_gemm(const u16* __restrict__ A, const u16* __restrict__ Bt,
            float* __restrict__ C, int R, int K, int N,
            size_t strideA, size_t strideB, size_t strideC) {
  const int lane = threadIdx.x & 31;
  const int c0 = blockIdx.x * 32;
  const int r0 = blockIdx.y * 64;
  A  += (size_t)blockIdx.z * strideA;
  Bt += (size_t)blockIdx.z * strideB;
  C  += (size_t)blockIdx.z * strideC;

  const u16* ap[4];
#pragma unroll
  for (int rt = 0; rt < 4; ++rt)
    ap[rt] = A + (size_t)(r0 + rt * 16 + (lane & 15)) * K + ((lane >> 4) * 16);
  const u16* bp0 = Bt + (size_t)(c0 + (lane & 15)) * K + ((lane >> 4) * 16);
  const u16* bp1 = bp0 + (size_t)16 * K;

  v8f acc[4][2];
#pragma unroll
  for (int rt = 0; rt < 4; ++rt) { acc[rt][0] = v8f{}; acc[rt][1] = v8f{}; }

  for (int k0 = 0; k0 < K; k0 += 32) {
    __builtin_prefetch(bp0 + k0 + 256, 0, 1);   // global_prefetch_b8
    Frag b0, b1;
    b0.q[0] = *(const Q4*)(bp0 + k0);
    b0.q[1] = *(const Q4*)(bp0 + k0 + 8);
    b1.q[0] = *(const Q4*)(bp1 + k0);
    b1.q[1] = *(const Q4*)(bp1 + k0 + 8);
#pragma unroll
    for (int rt = 0; rt < 4; ++rt) {
      Frag a;
      a.q[0] = *(const Q4*)(ap[rt] + k0);
      a.q[1] = *(const Q4*)(ap[rt] + k0 + 8);
      acc[rt][0] = __builtin_amdgcn_wmma_f32_16x16x32_bf16(
          false, a.v, false, b0.v, (short)0, acc[rt][0], false, false);
      acc[rt][1] = __builtin_amdgcn_wmma_f32_16x16x32_bf16(
          false, a.v, false, b1.v, (short)0, acc[rt][1], false, false);
    }
  }
  // C/D layout: VGPR v -> row base + (lane>=16 ? 8 : 0) + v, col c0 + lane%16
  const int ncol = c0 + (lane & 15);
#pragma unroll
  for (int rt = 0; rt < 4; ++rt) {
    const int rbase = r0 + rt * 16 + ((lane >> 4) * 8);
#pragma unroll
    for (int v = 0; v < 8; ++v) {
      C[(size_t)(rbase + v) * N + ncol]      = acc[rt][0][v];
      C[(size_t)(rbase + v) * N + ncol + 16] = acc[rt][1][v];
    }
  }
}

// ---------------------------------------------------------------------------
// GAT attention pieces (per meta-path m, node n, head h)
__global__ void k_el_er(const float* __restrict__ feat, const float* __restrict__ al,
                        const float* __restrict__ ar, float* __restrict__ el,
                        float* __restrict__ er, int H, int HD) {
  size_t i = (size_t)blockIdx.x * blockDim.x + threadIdx.x;
  size_t total = (size_t)MM * ND * H;
  if (i >= total) return;
  int h = (int)(i % H); size_t t = i / H;
  int n = (int)(t % ND); int m = (int)(t / ND);
  const float* f = feat + ((size_t)m * ND + n) * HD + h * DHEAD;
  const float* a = al + ((size_t)m * H + h) * DHEAD;
  const float* b = ar + ((size_t)m * H + h) * DHEAD;
  float sl = 0.f, sr = 0.f;
  for (int d = 0; d < DHEAD; ++d) { float x = f[d]; sl += x * a[d]; sr += x * b[d]; }
  el[i] = sl; er[i] = sr;
}

DEV float leaky02(float v) { return v > 0.f ? v : 0.2f * v; }

__global__ void k_edge_max(const int* __restrict__ src, const int* __restrict__ dst,
                           const float* __restrict__ el, const float* __restrict__ er,
                           float* __restrict__ emax, int H) {
  size_t i = (size_t)blockIdx.x * blockDim.x + threadIdx.x;
  size_t total = (size_t)MM * EDRUG * H;
  if (i >= total) return;
  int h = (int)(i % H); size_t t = i / H;
  int e = (int)(t % EDRUG); int m = (int)(t / EDRUG);
  int s = src[e], d = dst[e];
  float v = leaky02(el[((size_t)m * ND + s) * H + h] + er[((size_t)m * ND + d) * H + h]);
  atomicMaxF(&emax[((size_t)m * ND + d) * H + h], v);
}

__global__ void k_edge_exp(const int* __restrict__ src, const int* __restrict__ dst,
                           const float* __restrict__ el, const float* __restrict__ er,
                           const float* __restrict__ emax, float* __restrict__ ex,
                           float* __restrict__ den, int H) {
  size_t i = (size_t)blockIdx.x * blockDim.x + threadIdx.x;
  size_t total = (size_t)MM * EDRUG * H;
  if (i >= total) return;
  int h = (int)(i % H); size_t t = i / H;
  int e = (int)(t % EDRUG); int m = (int)(t / EDRUG);
  int s = src[e], d = dst[e];
  size_t di = ((size_t)m * ND + d) * H + h;
  float v = leaky02(el[((size_t)m * ND + s) * H + h] + er[di]);
  float x = __expf(v - emax[di]);
  ex[i] = x;
  atomicAdd(&den[di], x);
}

__global__ void k_edge_agg(const int* __restrict__ src, const int* __restrict__ dst,
                           const float* __restrict__ ex, const float* __restrict__ den,
                           const float* __restrict__ feat, float* __restrict__ agg,
                           int H, int HD) {
  size_t i = (size_t)blockIdx.x * blockDim.x + threadIdx.x;
  size_t total = (size_t)MM * EDRUG * H;
  if (i >= total) return;
  int h = (int)(i % H); size_t t = i / H;
  int e = (int)(t % EDRUG); int m = (int)(t / EDRUG);
  int s = src[e], d = dst[e];
  float alpha = ex[i] / den[((size_t)m * ND + d) * H + h];
  const float* fs = feat + ((size_t)m * ND + s) * HD + h * DHEAD;
  float* ad = agg + ((size_t)m * ND + d) * HD + h * DHEAD;
  for (int dd = 0; dd < DHEAD; ++dd) atomicAdd(&ad[dd], alpha * fs[dd]);
}

__global__ void k_bias_elu(float* __restrict__ agg, const float* __restrict__ bg, int HD) {
  size_t i = (size_t)blockIdx.x * blockDim.x + threadIdx.x;
  size_t total = (size_t)MM * ND * HD;
  if (i >= total) return;
  int j = (int)(i % HD); int m = (int)(i / ((size_t)ND * HD));
  float x = agg[i] + bg[(size_t)m * HD + j];
  agg[i] = x > 0.f ? x : expm1f(x);
}

// ---------------------------------------------------------------------------
// Semantic attention
__global__ void k_sem_logits(const float* __restrict__ t, const float* __restrict__ sb1,
                             const float* __restrict__ sw2, float* __restrict__ w,
                             size_t rows) {
  size_t i = (size_t)blockIdx.x * blockDim.x + threadIdx.x;
  if (i >= rows) return;
  const float* tr = t + i * 128;
  float s = 0.f;
  for (int k = 0; k < 128; ++k) s += tanhf(tr[k] + sb1[k]) * sw2[k];
  w[i] = s;
}

__global__ void k_sem_softmax(float* __restrict__ w) {
  int n = blockIdx.x * blockDim.x + threadIdx.x;
  if (n >= ND) return;
  float mx = -3.0e38f;
  for (int m = 0; m < MM; ++m) mx = fmaxf(mx, w[(size_t)m * ND + n]);
  float s = 0.f;
  for (int m = 0; m < MM; ++m) s += __expf(w[(size_t)m * ND + n] - mx);
  for (int m = 0; m < MM; ++m)
    w[(size_t)m * ND + n] = __expf(w[(size_t)m * ND + n] - mx) / s;
}

__global__ void k_sem_combine(const float* __restrict__ w, const float* __restrict__ z,
                              float* __restrict__ xnext, int HD) {
  size_t i = (size_t)blockIdx.x * blockDim.x + threadIdx.x;
  size_t total = (size_t)ND * HD;
  if (i >= total) return;
  int n = (int)(i / HD); int j = (int)(i % HD);
  float s = 0.f;
  for (int m = 0; m < MM; ++m)
    s += w[(size_t)m * ND + n] * z[((size_t)m * ND + n) * HD + j];
  xnext[i] = s;
}

// ---------------------------------------------------------------------------
// Generic small f32 GEMM: C = act(A@W + bias + resid); act 0=none,1=relu,2=tanh
__global__ void k_sgemm(const float* __restrict__ A, const float* __restrict__ W,
                        const float* __restrict__ bias, const float* __restrict__ resid,
                        float* __restrict__ C, int R, int K, int N, int act) {
  size_t i = (size_t)blockIdx.x * blockDim.x + threadIdx.x;
  if (i >= (size_t)R * N) return;
  int r = (int)(i / N), n = (int)(i - (size_t)r * N);
  float s = bias ? bias[n] : 0.f;
  const float* a = A + (size_t)r * K;
  for (int k = 0; k < K; ++k) s += a[k] * W[(size_t)k * N + n];
  if (resid) s += resid[i];
  if (act == 1) s = fmaxf(s, 0.f);
  else if (act == 2) s = tanhf(s);
  C[i] = s;
}

// out[b,j] = max_{i<segRows} x[(b*segStride+i)*D + j]
__global__ void k_segmax(const float* __restrict__ x, float* __restrict__ out,
                         int segRows, int segStride, int D) {
  int i = blockIdx.x * blockDim.x + threadIdx.x;
  if (i >= BB * D) return;
  int b = i / D, j = i - b * D;
  float mx = -3.0e38f;
  const float* p = x + (size_t)b * segStride * D + j;
  for (int r = 0; r < segRows; ++r) mx = fmaxf(mx, p[(size_t)r * D]);
  out[i] = mx;
}

// ---------------------------------------------------------------------------
// Protein branch helpers
__global__ void k_prot_logits(const float* __restrict__ tres, const float* __restrict__ slw,
                              const float* __restrict__ slb, float* __restrict__ logits) {
  int i = blockIdx.x * blockDim.x + threadIdx.x;
  if (i >= BB * LL) return;
  int b = i / LL, l = i - b * LL;
  const float* row = tres + ((size_t)b * PSTRIDE + l) * 64;
  float s = slb[0];
  for (int k = 0; k < 64; ++k) s += row[k] * slw[k];
  logits[i] = s;
}

__global__ void k_softmax_rows(const float* __restrict__ logits, float* __restrict__ natt) {
  __shared__ float red[256];
  int b = blockIdx.x, t = threadIdx.x;
  const float* lg = logits + (size_t)b * LL;
  float mx = -3.0e38f;
  for (int i = t; i < LL; i += 256) mx = fmaxf(mx, lg[i]);
  red[t] = mx; __syncthreads();
  for (int s = 128; s > 0; s >>= 1) { if (t < s) red[t] = fmaxf(red[t], red[t + s]); __syncthreads(); }
  mx = red[0]; __syncthreads();
  float sum = 0.f;
  for (int i = t; i < LL; i += 256) sum += __expf(lg[i] - mx);
  red[t] = sum; __syncthreads();
  for (int s = 128; s > 0; s >>= 1) { if (t < s) red[t] += red[t + s]; __syncthreads(); }
  sum = red[0];
  for (int i = t; i < LL; i += 256) natt[(size_t)b * LL + i] = __expf(lg[i] - mx) / sum;
}

__global__ void k_set_rows(float* __restrict__ x2, const float* __restrict__ xfc) {
  int i = blockIdx.x * blockDim.x + threadIdx.x;
  if (i >= BB * 64) return;
  int b = i / 64, j = i - b * 64;
  x2[((size_t)b * PSTRIDE + LL) * 64 + j] = xfc[i];
}

__global__ void k_gather_rows(const float* __restrict__ x2, float* __restrict__ dra) {
  int i = blockIdx.x * blockDim.x + threadIdx.x;
  if (i >= BB * 64) return;
  int b = i / 64, j = i - b * 64;
  dra[i] = x2[((size_t)b * PSTRIDE + LL) * 64 + j];
}

__global__ void k_ea_write(const float* __restrict__ natt, float* __restrict__ ea) {
  int i = blockIdx.x * blockDim.x + threadIdx.x;
  if (i >= BB * 2 * LL) return;
  int b = i / (2 * LL), l = i % LL;
  ea[i] = natt[(size_t)b * LL + l];
}

__global__ void k_deg_edges(const int* __restrict__ dst, const float* __restrict__ w,
                            float* __restrict__ deg, int E) {
  int e = blockIdx.x * blockDim.x + threadIdx.x;
  if (e >= E) return;
  atomicAdd(&deg[dst[e]], w ? w[e] : 1.0f);
}

__global__ void k_rsqrt(const float* __restrict__ deg, float* __restrict__ dis, int n) {
  int i = blockIdx.x * blockDim.x + threadIdx.x;
  if (i >= n) return;
  float d = deg[i];
  dis[i] = d > 0.f ? rsqrtf(d) : 0.f;
}

__global__ void k_gcn_init(const float* __restrict__ h, const float* __restrict__ dis,
                           const float* __restrict__ bias, float* __restrict__ out, int D) {
  size_t i = (size_t)blockIdx.x * blockDim.x + threadIdx.x;
  if (i >= (size_t)NPROT * D) return;
  int n = (int)(i / D), j = (int)(i % D);
  out[i] = dis[n] * dis[n] * h[i] + bias[j];   // self-loop (w=1) + bias
}

__global__ void k_gcn_agg(const int* __restrict__ src, const int* __restrict__ dst,
                          const float* __restrict__ w, const float* __restrict__ dis,
                          const float* __restrict__ h, float* __restrict__ out,
                          int E, int D) {
  int e = blockIdx.x * blockDim.x + threadIdx.x;
  if (e >= E) return;
  int s = src[e], d = dst[e];
  float norm = dis[s] * (w ? w[e] : 1.0f) * dis[d];
  if (norm == 0.f) return;
  const float* hs = h + (size_t)s * D;
  float* od = out + (size_t)d * D;
  for (int j = 0; j < D; ++j) atomicAdd(&od[j], norm * hs[j]);
}

__global__ void k_relu(float* __restrict__ x, size_t n) {
  size_t i = (size_t)blockIdx.x * blockDim.x + threadIdx.x;
  if (i < n) x[i] = fmaxf(x[i], 0.f);
}

__global__ void k_maxpair(const float* __restrict__ dra, const float* __restrict__ xfc,
                          float* __restrict__ xp) {
  int i = blockIdx.x * blockDim.x + threadIdx.x;
  if (i >= BB * 64) return;
  xp[i] = fmaxf(dra[i], fmaxf(xfc[i], 0.f));
}

__global__ void k_concat(const float* __restrict__ xp, const float* __restrict__ tt,
                         float* __restrict__ xc) {
  int i = blockIdx.x * blockDim.x + threadIdx.x;
  if (i >= BB * 128) return;
  int b = i / 128, j = i - b * 128;
  xc[i] = (j < 64) ? xp[b * 64 + j] : tt[b * 64 + (j - 64)];
}

// ===========================================================================
#define G1(n) dim3((unsigned)(((size_t)(n) + 255) / 256)), dim3(256), 0, stream

extern "C" void kernel_launch(void* const* d_in, const int* in_sizes, int n_in,
                              void* d_out, int out_size, void* d_ws, size_t ws_size,
                              hipStream_t stream) {
  (void)in_sizes; (void)n_in; (void)out_size; (void)ws_size;
  auto F = [&](int i) { return (const float*)d_in[i]; };
  const int* dsrc = (const int*)d_in[1];
  const int* ddst = dsrc + EDRUG;
  const int* psrc = (const int*)d_in[4];
  const int* pdst = psrc + EPROT;

  // ---- workspace bump allocator ------------------------------------------
  char* ws = (char*)d_ws;
  size_t off = 0;
  auto alloc = [&](size_t bytes) -> void* {
    void* p = ws + off;
    off = (off + bytes + 255) & ~(size_t)255;
    return p;
  };
  float* XD   = (float*)alloc((size_t)ND * 768 * 4);        // drug activation
  u16*   XB   = (u16*)  alloc((size_t)ND * 768 * 2);        // bf16 activation
  u16*   WT   = (u16*)  alloc((size_t)MM * 768 * 768 * 2);  // bf16 W^T (reused for sw1^T)
  float* FEAT = (float*)alloc((size_t)MM * ND * 768 * 4);   // [M,N,HD]
  float* AGG  = (float*)alloc((size_t)MM * ND * 768 * 4);   // GAT out -> z
  u16*   ZB   = (u16*)FEAT;                                  // alias: FEAT dead by then
  float* T1   = (float*)((char*)FEAT + (size_t)MM * ND * 768 * 2); // alias upper half
  float* EL   = (float*)alloc((size_t)MM * ND * 12 * 4);
  float* ER   = (float*)alloc((size_t)MM * ND * 12 * 4);
  float* EMAX = (float*)alloc((size_t)MM * ND * 12 * 4);
  float* DEN  = (float*)alloc((size_t)MM * ND * 12 * 4);
  float* EX   = (float*)alloc((size_t)MM * EDRUG * 12 * 4);
  float* WSEM = (float*)alloc((size_t)MM * ND * 4);
  float* XG   = (float*)alloc((size_t)BB * 384 * 4);
  float* XFC  = (float*)alloc((size_t)BB * 64 * 4);
  float* TRES = (float*)alloc((size_t)NPROT * 64 * 4);
  float* LOGI = (float*)alloc((size_t)BB * LL * 4);
  float* NATT = (float*)alloc((size_t)BB * LL * 4);
  float* X2   = (float*)alloc((size_t)NPROT * 64 * 4);
  float* HA   = (float*)alloc((size_t)NPROT * 64 * 4);
  float* HB   = (float*)alloc((size_t)NPROT * 64 * 4);
  float* DEG  = (float*)alloc((size_t)NPROT * 4);
  float* DIS  = (float*)alloc((size_t)NPROT * 4);
  float* DEG2 = (float*)alloc((size_t)NPROT * 4);
  float* DIS2 = (float*)alloc((size_t)NPROT * 4);
  float* EA   = (float*)alloc((size_t)EPROT * 4);
  float* DRA  = (float*)alloc((size_t)BB * 64 * 4);
  float* XG2  = (float*)alloc((size_t)BB * 64 * 4);
  float* T1T  = (float*)alloc((size_t)BB * 1024 * 4);
  float* TT   = (float*)alloc((size_t)BB * 64 * 4);
  float* XP   = (float*)alloc((size_t)BB * 64 * 4);
  float* XC   = (float*)alloc((size_t)BB * 128 * 4);
  float* FB1  = (float*)alloc((size_t)BB * 1024 * 4);
  float* FB2  = (float*)alloc((size_t)BB * 512 * 4);

  // ---- drug branch: 4 HGAN layers ----------------------------------------
  const int INS_[4]   = {64, 768, 640, 512};
  const int HEADS_[4] = {12, 10, 8, 6};
  for (int l = 0; l < 4; ++l) {
    const int IN = INS_[l], H = HEADS_[l], HD = H * DHEAD;
    const int pb = 23 + 7 * l;
    const float* Wg  = F(pb + 0); const float* al  = F(pb + 1);
    const float* ar  = F(pb + 2); const float* bg  = F(pb + 3);
    const float* sb1 = F(pb + 4); const float* sw1 = F(pb + 5);
    const float* sw2 = F(pb + 6);
    const float* xin = (l == 0) ? F(0) : XD;

    size_t nx = (size_t)ND * IN;
    k_f2bf<<<G1(nx)>>>(xin, XB, nx);
    size_t wn = (size_t)IN * HD;
    k_transpose_bf<<<dim3((unsigned)((wn + 255) / 256), 1, MM), dim3(256), 0, stream>>>(
        Wg, WT, IN, HD);
    // feat[m] = X @ Wg[m]   (A shared across batch: strideA = 0)
    k_wmma_gemm<<<dim3(HD / 32, ND / 64, MM), dim3(32), 0, stream>>>(
        XB, WT, FEAT, ND, IN, HD, (size_t)0, wn, (size_t)ND * HD);

    size_t nh = (size_t)MM * ND * H;
    k_el_er<<<G1(nh)>>>(FEAT, al, ar, EL, ER, H, HD);
    k_fill<<<G1(nh)>>>(EMAX, -3.0e38f, nh);
    k_fill<<<G1(nh)>>>(DEN, 0.0f, nh);
    size_t nagg = (size_t)MM * ND * HD;
    k_fill<<<G1(nagg)>>>(AGG, 0.0f, nagg);
    size_t neh = (size_t)MM * EDRUG * H;
    k_edge_max<<<G1(neh)>>>(dsrc, ddst, EL, ER, EMAX, H);
    k_edge_exp<<<G1(neh)>>>(dsrc, ddst, EL, ER, EMAX, EX, DEN, H);
    k_edge_agg<<<G1(neh)>>>(dsrc, ddst, EX, DEN, FEAT, AGG, H, HD);
    k_bias_elu<<<G1(nagg)>>>(AGG, bg, HD);

    // semantic attention: t = z @ sw1  (z = AGG as [M*N, HD] bf16)
    k_f2bf<<<G1(nagg)>>>(AGG, ZB, nagg);
    size_t sn = (size_t)HD * 128;
    k_transpose_bf<<<dim3((unsigned)((sn + 255) / 256), 1, 1), dim3(256), 0, stream>>>(
        sw1, WT, HD, 128);
    k_wmma_gemm<<<dim3(128 / 32, (MM * ND) / 64, 1), dim3(32), 0, stream>>>(
        ZB, WT, T1, MM * ND, HD, 128, (size_t)0, (size_t)0, (size_t)0);
    size_t rows = (size_t)MM * ND;
    k_sem_logits<<<G1(rows)>>>(T1, sb1, sw2, WSEM, rows);
    k_sem_softmax<<<G1(ND)>>>(WSEM);
    size_t nc = (size_t)ND * HD;
    k_sem_combine<<<G1(nc)>>>(WSEM, AGG, XD, HD);
  }

  // ---- drug pooling + projection -----------------------------------------
  k_segmax<<<G1(BB * 384)>>>(XD, XG, NATM, NATM, 384);
  k_sgemm<<<G1(BB * 64)>>>(XG, F(16), F(15), nullptr, XFC, BB, 384, 64, 0);

  // ---- protein node attention --------------------------------------------
  k_sgemm<<<G1((size_t)NPROT * 64)>>>(F(3), F(22), F(21), nullptr, TRES, NPROT, 64, 64, 2);
  k_prot_logits<<<G1(BB * LL)>>>(TRES, F(57), F(56), LOGI);
  k_softmax_rows<<<dim3(BB), dim3(256), 0, stream>>>(LOGI, NATT);

  // ---- build x2 and edge weights -----------------------------------------
  hipMemcpyAsync(X2, F(3), (size_t)NPROT * 64 * sizeof(float),
                 hipMemcpyDeviceToDevice, stream);
  k_set_rows<<<G1(BB * 64)>>>(X2, XFC);
  k_fill<<<G1(EPROT)>>>(EA, 1.0f, (size_t)EPROT);
  k_ea_write<<<G1(BB * 2 * LL)>>>(NATT, EA);

  // ---- GCN c1 / c2 (edge-weighted) ---------------------------------------
  k_fill<<<G1(NPROT)>>>(DEG, 1.0f, (size_t)NPROT);   // self loops
  k_deg_edges<<<G1(EPROT)>>>(pdst, EA, DEG, EPROT);
  k_rsqrt<<<G1(NPROT)>>>(DEG, DIS, NPROT);

  k_sgemm<<<G1((size_t)NPROT * 32)>>>(X2, F(8), nullptr, nullptr, HA, NPROT, 64, 32, 0);
  k_gcn_init<<<G1((size_t)NPROT * 32)>>>(HA, DIS, F(7), HB, 32);
  k_gcn_agg<<<G1(EPROT)>>>(psrc, pdst, EA, DIS, HA, HB, EPROT, 32);
  k_relu<<<G1((size_t)NPROT * 32)>>>(HB, (size_t)NPROT * 32);

  k_sgemm<<<G1((size_t)NPROT * 64)>>>(HB, F(10), nullptr, nullptr, HA, NPROT, 32, 64, 0);
  k_gcn_init<<<G1((size_t)NPROT * 64)>>>(HA, DIS, F(9), X2, 64);
  k_gcn_agg<<<G1(EPROT)>>>(psrc, pdst, EA, DIS, HA, X2, EPROT, 64);
  k_relu<<<G1((size_t)NPROT * 64)>>>(X2, (size_t)NPROT * 64);

  // ---- 4x residual GCN blocks (unit edge weights, shared weights) --------
  k_fill<<<G1(NPROT)>>>(DEG2, 1.0f, (size_t)NPROT);
  k_deg_edges<<<G1(EPROT)>>>(pdst, nullptr, DEG2, EPROT);
  k_rsqrt<<<G1(NPROT)>>>(DEG2, DIS2, NPROT);
  for (int r = 0; r < 4; ++r) {
    k_sgemm<<<G1((size_t)NPROT * 64)>>>(X2, F(54), nullptr, nullptr, HA, NPROT, 64, 64, 0);
    k_gcn_init<<<G1((size_t)NPROT * 64)>>>(HA, DIS2, F(53), HB, 64);
    k_gcn_agg<<<G1(EPROT)>>>(psrc, pdst, nullptr, DIS2, HA, HB, EPROT, 64);
    k_relu<<<G1((size_t)NPROT * 64)>>>(HB, (size_t)NPROT * 64);
    k_sgemm<<<G1((size_t)NPROT * 64)>>>(HB, F(55), nullptr, X2, X2, NPROT, 64, 64, 1);
  }

  // ---- tail ---------------------------------------------------------------
  k_gather_rows<<<G1(BB * 64)>>>(X2, DRA);
  k_segmax<<<G1(BB * 64)>>>(X2, XG2, LL, PSTRIDE, 64);
  k_sgemm<<<G1(BB * 1024)>>>(XG2, F(18), F(17), nullptr, T1T, BB, 64, 1024, 1);
  k_sgemm<<<G1(BB * 64)>>>(T1T, F(20), F(19), nullptr, TT, BB, 1024, 64, 0);
  k_maxpair<<<G1(BB * 64)>>>(DRA, XFC, XP);
  k_concat<<<G1(BB * 128)>>>(XP, TT, XC);
  k_sgemm<<<G1(BB * 1024)>>>(XC, F(12), F(11), nullptr, FB1, BB, 128, 1024, 1);
  k_sgemm<<<G1(BB * 512)>>>(FB1, F(14), F(13), nullptr, FB2, BB, 1024, 512, 1);
  k_sgemm<<<G1(BB)>>>(FB2, F(52), F(51), nullptr, (float*)d_out, BB, 512, 1, 0);
}
